// BeansAttentionBlock_14010183320078
// MI455X (gfx1250) — matrix operand order repaired and checked by reading, not compiled, not run on hardware
//
#include <hip/hip_runtime.h>
#include <math.h>

// ---------------- problem constants ----------------
constexpr int Bc  = 4;
constexpr int Pc  = 1024;
constexpr int KNB = 32;
constexpr int Dc  = 768;
constexpr int Hc  = 12;
constexpr int Sc  = Pc + 1;        // 1025
constexpr int HDc = 64;
constexpr int Mrows = Bc * Sc;     // 4100

typedef unsigned short u16;
typedef unsigned int   u32;
typedef __attribute__((ext_vector_type(16))) __bf16 v16bf;
typedef __attribute__((ext_vector_type(8)))  float  v8f;

__device__ __forceinline__ float bf2f(u16 u) {
    return __uint_as_float(((u32)u) << 16);
}
__device__ __forceinline__ u16 f2bf(float f) {            // round-to-nearest-even
    u32 u = __float_as_uint(f);
    return (u16)((u + 0x7FFFu + ((u >> 16) & 1u)) >> 16);
}

// ---------------- fp32 (K x N) -> bf16 transposed (N x K) ----------------
__global__ __launch_bounds__(256)
void transpose_bf16_kernel(const float* __restrict__ in, u16* __restrict__ out,
                           int Kdim, int Ndim) {
    __shared__ float tile[32][33];
    int bx = blockIdx.x * 32;           // n base
    int by = blockIdx.y * 32;           // k base
    int tx = threadIdx.x & 31;
    int ty = threadIdx.x >> 5;          // 0..7
    #pragma unroll
    for (int i = 0; i < 4; i++)
        tile[ty + 8 * i][tx] = in[(size_t)(by + ty + 8 * i) * Ndim + bx + tx];
    __syncthreads();
    #pragma unroll
    for (int i = 0; i < 4; i++)
        out[(size_t)(bx + ty + 8 * i) * Kdim + by + tx] = f2bf(tile[tx][ty + 8 * i]);
}

// ---------------- LayerNorm (row of D=768) -> bf16 ----------------
__global__ __launch_bounds__(256)
void ln_kernel(const float* __restrict__ x, const float* __restrict__ w,
               const float* __restrict__ b, u16* __restrict__ out) {
    int row = blockIdx.x;
    int tid = threadIdx.x;
    const float* xr = x + (size_t)row * Dc;
    float v0 = xr[tid], v1 = xr[tid + 256], v2 = xr[tid + 512];
    float s  = v0 + v1 + v2;
    float s2 = v0 * v0 + v1 * v1 + v2 * v2;
    for (int o = 16; o > 0; o >>= 1) { s += __shfl_down(s, o); s2 += __shfl_down(s2, o); }
    __shared__ float rs[8], rs2[8];
    int lane = tid & 31, wv = tid >> 5;
    if (lane == 0) { rs[wv] = s; rs2[wv] = s2; }
    __syncthreads();
    if (wv == 0) {
        float a  = (lane < 8) ? rs[lane]  : 0.f;
        float a2 = (lane < 8) ? rs2[lane] : 0.f;
        for (int o = 4; o > 0; o >>= 1) { a += __shfl_down(a, o); a2 += __shfl_down(a2, o); }
        if (lane == 0) { rs[0] = a; rs2[0] = a2; }
    }
    __syncthreads();
    float mean = rs[0] * (1.0f / Dc);
    float var  = rs2[0] * (1.0f / Dc) - mean * mean;
    float rstd = rsqrtf(var + 1e-5f);
    size_t o0 = (size_t)row * Dc;
    out[o0 + tid]       = f2bf((v0 - mean) * rstd * w[tid]       + b[tid]);
    out[o0 + tid + 256] = f2bf((v1 - mean) * rstd * w[tid + 256] + b[tid + 256]);
    out[o0 + tid + 512] = f2bf((v2 - mean) * rstd * w[tid + 512] + b[tid + 512]);
}

// ---------------- WMMA bf16 GEMM ----------------
// 128x128 block tile, 8 waves, each wave owns 32x64 (8 WMMA tiles).
// B is pre-transposed to [N][K] so both LDS stages are b128 row copies.
// Single-barrier LDS double buffering.
constexpr int EP_QKV   = 0;   // + bias, scatter bf16 into [3][B][H][S][HD]
constexpr int EP_RESID = 1;   // + bias + resid, fp32 row-major store
constexpr int EP_GELU  = 2;   // + bias, exact gelu, bf16 row-major store

constexpr int LSTR = 40;      // LDS row stride in u16 (conflict-friendly padding)

template<int EPI>
__global__ __launch_bounds__(256)
void gemm_bf16_kernel(const u16* __restrict__ A, const u16* __restrict__ Bt,
                      const float* __restrict__ bias, const float* __restrict__ resid,
                      float* __restrict__ outF, u16* __restrict__ outB,
                      int M, int N, int K) {
    __shared__ __align__(16) u16 As[2][128 * LSTR];   // [m][k]
    __shared__ __align__(16) u16 Bs[2][128 * LSTR];   // [n][k]

    const int tid  = threadIdx.x;
    const int lane = tid & 31;
    const int wave = tid >> 5;
    const int wm = wave & 3;          // 0..3 -> 32-row slice
    const int wn = wave >> 2;         // 0..1 -> 64-col slice
    const int mBase = blockIdx.y * 128;
    const int nBase = blockIdx.x * 128;

    v8f c[2][4] = {};

    // staging: each thread copies 16 contiguous bf16 of one row (2 threads/row)
    const int ldRow = tid >> 1;             // 0..127
    const int ldOff = (tid & 1) * 16;       // 0 or 16
    const bool aValid = (mBase + ldRow) < M;
    const u16* gA = A  + (size_t)(mBase + ldRow) * K + ldOff;
    const u16* gB = Bt + (size_t)(nBase + ldRow) * K + ldOff;
    const int lBase = ldRow * LSTR + ldOff;

    // fragment addressing (per ISA 7.12.2 layouts)
    const int am0 = (wm * 32 + (lane & 15)) * LSTR + ((lane < 16) ? 0 : 8);
    const int am1 = am0 + 16 * LSTR;
    const int bn0 = (wn * 64 + (lane & 15)) * LSTR + ((lane < 16) ? 0 : 8);

    const int nsteps = K / 32;

    // preload step 0
    uint4 pa0 = {0u,0u,0u,0u}, pa1 = {0u,0u,0u,0u};
    if (aValid) { pa0 = *(const uint4*)(gA); pa1 = *(const uint4*)(gA + 8); }
    uint4 pb0 = *(const uint4*)(gB);
    uint4 pb1 = *(const uint4*)(gB + 8);
    *(uint4*)(&As[0][lBase])     = pa0;
    *(uint4*)(&As[0][lBase + 8]) = pa1;
    *(uint4*)(&Bs[0][lBase])     = pb0;
    *(uint4*)(&Bs[0][lBase + 8]) = pb1;
    __syncthreads();

    for (int s = 0; s < nsteps; s++) {
        const int cur = s & 1;
        // prefetch next K-slice into registers
        if (s + 1 < nsteps) {
            const int k = (s + 1) * 32;
            pa0 = {0u,0u,0u,0u}; pa1 = {0u,0u,0u,0u};
            if (aValid) { pa0 = *(const uint4*)(gA + k); pa1 = *(const uint4*)(gA + k + 8); }
            pb0 = *(const uint4*)(gB + k);
            pb1 = *(const uint4*)(gB + k + 8);
        }

        // fragments from buffer `cur`
        union F { v16bf v; uint4 q[2]; };
        F af[2], bfm[4];
        af[0].q[0] = *(const uint4*)(&As[cur][am0]);
        af[0].q[1] = *(const uint4*)(&As[cur][am0 + 16]);
        af[1].q[0] = *(const uint4*)(&As[cur][am1]);
        af[1].q[1] = *(const uint4*)(&As[cur][am1 + 16]);
        #pragma unroll
        for (int j = 0; j < 4; j++) {
            bfm[j].q[0] = *(const uint4*)(&Bs[cur][bn0 + j * 16 * LSTR]);
            bfm[j].q[1] = *(const uint4*)(&Bs[cur][bn0 + j * 16 * LSTR + 16]);
        }
        #pragma unroll
        for (int i = 0; i < 2; i++)
            #pragma unroll
            for (int j = 0; j < 4; j++)
                c[i][j] = __builtin_amdgcn_wmma_f32_16x16x32_bf16(
                    false, af[i].v, false, bfm[j].v, (short)0, c[i][j], false, false);

        // stage next tile into the other buffer
        if (s + 1 < nsteps) {
            const int nxt = 1 - cur;
            *(uint4*)(&As[nxt][lBase])     = pa0;
            *(uint4*)(&As[nxt][lBase + 8]) = pa1;
            *(uint4*)(&Bs[nxt][lBase])     = pb0;
            *(uint4*)(&Bs[nxt][lBase + 8]) = pb1;
        }
        __syncthreads();
    }

    // epilogue: 8 tiles of 16x16 per wave
    #pragma unroll
    for (int i = 0; i < 2; i++) {
        #pragma unroll
        for (int j = 0; j < 4; j++) {
            int n = nBase + wn * 64 + j * 16 + (lane & 15);
            float bsv = bias[n];
            int mB = mBase + wm * 32 + i * 16 + ((lane >= 16) ? 8 : 0);
            #pragma unroll
            for (int v = 0; v < 8; v++) {
                int m = mB + v;
                if (m >= M) continue;
                float val = c[i][j][v] + bsv;
                if (EPI == EP_RESID) {
                    val += resid[(size_t)m * N + n];
                    outF[(size_t)m * N + n] = val;
                } else if (EPI == EP_GELU) {
                    float g = 0.5f * val * (1.0f + erff(val * 0.70710678118654752f));
                    outB[(size_t)m * N + n] = f2bf(g);
                } else { // EP_QKV scatter -> [3][B][H][S][HD]
                    int which = n / Dc;
                    int rem   = n - which * Dc;
                    int h = rem >> 6;
                    int d = rem & 63;
                    int b = m / Sc;
                    int ss = m - b * Sc;
                    size_t idx = ((((size_t)which * Bc + b) * Hc + h) * Sc + ss) * HDc + d;
                    outB[idx] = f2bf(val);
                }
            }
        }
    }
}

// ---------------- routed attention: one wave per (b,h,p), one lane per key ----------------
__global__ __launch_bounds__(256)
void routed_attn_kernel(const u16* __restrict__ qkv, const int* __restrict__ routes,
                        u16* __restrict__ out) {
    int gid  = blockIdx.x * 8 + (threadIdx.x >> 5);
    int lane = threadIdx.x & 31;
    if (gid >= Bc * Hc * Pc) return;
    int p = gid % Pc;
    int t = gid / Pc;
    int h = t % Hc;
    int b = t / Hc;

    const u16* Qb = qkv + (((size_t)0 * Bc + b) * Hc + h) * (size_t)Sc * HDc;
    const u16* Kb = qkv + (((size_t)1 * Bc + b) * Hc + h) * (size_t)Sc * HDc;
    const u16* Vb = qkv + (((size_t)2 * Bc + b) * Hc + h) * (size_t)Sc * HDc;

    const u16* q = Qb + (size_t)(p + 1) * HDc;
    int r = routes[p * KNB + lane] + 1;
    const u16* k = Kb + (size_t)r * HDc;

    float s = 0.f;
    #pragma unroll
    for (int i = 0; i < 8; i++) {
        uint4 qv = *(const uint4*)(q + i * 8);
        uint4 kv = *(const uint4*)(k + i * 8);
        const u16* qe = (const u16*)&qv;
        const u16* ke = (const u16*)&kv;
        #pragma unroll
        for (int j = 0; j < 8; j++) s += bf2f(qe[j]) * bf2f(ke[j]);
    }
    s *= 0.125f;  // HD^-0.5

    float m = s;
    for (int o = 16; o > 0; o >>= 1) m = fmaxf(m, __shfl_xor(m, o));
    float e = __expf(s - m);
    float sum = e;
    for (int o = 16; o > 0; o >>= 1) sum += __shfl_xor(sum, o);
    float w = e / sum;

    float acc0 = 0.f, acc1 = 0.f;
    #pragma unroll 4
    for (int kk = 0; kk < 32; kk++) {
        float wk = __shfl(w, kk);
        int   rk = __shfl(r, kk);
        const u16* vr = Vb + (size_t)rk * HDc;
        acc0 += wk * bf2f(vr[lane]);
        acc1 += wk * bf2f(vr[lane + 32]);
    }
    size_t orow = ((size_t)b * Sc + (p + 1)) * Dc + h * HDc;
    out[orow + lane]      = f2bf(acc0);
    out[orow + lane + 32] = f2bf(acc1);
}

// ---------------- cls attention: one block per (b,h), full S keys ----------------
__global__ __launch_bounds__(256)
void cls_attn_kernel(const u16* __restrict__ qkv, u16* __restrict__ out) {
    int h = blockIdx.x % Hc;
    int b = blockIdx.x / Hc;
    const u16* Qb = qkv + (((size_t)0 * Bc + b) * Hc + h) * (size_t)Sc * HDc;
    const u16* Kb = qkv + (((size_t)1 * Bc + b) * Hc + h) * (size_t)Sc * HDc;
    const u16* Vb = qkv + (((size_t)2 * Bc + b) * Hc + h) * (size_t)Sc * HDc;

    __shared__ float qs[64];
    __shared__ float sc[Sc + 3];
    __shared__ float red[256];
    int tid = threadIdx.x;
    if (tid < 64) qs[tid] = bf2f(Qb[tid]);
    __syncthreads();

    float lmax = -1e30f;
    for (int s0 = tid; s0 < Sc; s0 += 256) {
        const u16* kr = Kb + (size_t)s0 * HDc;
        float s = 0.f;
        #pragma unroll 8
        for (int d = 0; d < 64; d++) s += qs[d] * bf2f(kr[d]);
        s *= 0.125f;
        sc[s0] = s;
        lmax = fmaxf(lmax, s);
    }
    red[tid] = lmax; __syncthreads();
    for (int o = 128; o > 0; o >>= 1) { if (tid < o) red[tid] = fmaxf(red[tid], red[tid + o]); __syncthreads(); }
    float m = red[0];
    __syncthreads();

    float lsum = 0.f;
    for (int s0 = tid; s0 < Sc; s0 += 256) { float e = __expf(sc[s0] - m); sc[s0] = e; lsum += e; }
    red[tid] = lsum; __syncthreads();
    for (int o = 128; o > 0; o >>= 1) { if (tid < o) red[tid] += red[tid + o]; __syncthreads(); }
    float inv = 1.f / red[0];
    __syncthreads();

    int d = tid & 63;
    int g = tid >> 6;   // 0..3
    float acc = 0.f;
    for (int s0 = g; s0 < Sc; s0 += 4) acc += sc[s0] * bf2f(Vb[(size_t)s0 * HDc + d]);
    red[tid] = acc; __syncthreads();
    if (g == 0) {
        float a = red[d] + red[64 + d] + red[128 + d] + red[192 + d];
        out[((size_t)b * Sc) * Dc + h * HDc + d] = f2bf(a * inv);
    }
}

// ---------------- host-side orchestration ----------------
extern "C" void kernel_launch(void* const* d_in, const int* in_sizes, int n_in,
                              void* d_out, int out_size, void* d_ws, size_t ws_size,
                              hipStream_t stream) {
    (void)in_sizes; (void)n_in; (void)out_size; (void)ws_size;
    const float* x      = (const float*)d_in[0];
    const int*   routes = (const int*)d_in[1];
    const float* qkv_w  = (const float*)d_in[2];
    const float* qkv_b  = (const float*)d_in[3];
    const float* proj_w = (const float*)d_in[4];
    const float* proj_b = (const float*)d_in[5];
    const float* ln1_w  = (const float*)d_in[6];
    const float* ln1_b  = (const float*)d_in[7];
    const float* ln2_w  = (const float*)d_in[8];
    const float* ln2_b  = (const float*)d_in[9];
    const float* mlp_w1 = (const float*)d_in[10];
    const float* mlp_b1 = (const float*)d_in[11];
    const float* mlp_w2 = (const float*)d_in[12];
    const float* mlp_b2 = (const float*)d_in[13];
    float* out = (float*)d_out;

    char* wsp = (char*)d_ws;
    auto carve = [&](size_t bytes) -> char* {
        char* p = wsp;
        wsp += (bytes + 255) & ~(size_t)255;
        return p;
    };
    // weights stored transposed: [N][K] bf16
    u16* qkvw_t  = (u16*)carve((size_t)3 * Dc * Dc * 2);        // 2304 x 768
    u16* projw_t = (u16*)carve((size_t)Dc * Dc * 2);            // 768 x 768
    u16* w1_t    = (u16*)carve((size_t)4 * Dc * Dc * 2);        // 3072 x 768
    u16* w2_t    = (u16*)carve((size_t)Dc * 4 * Dc * 2);        // 768 x 3072
    u16* xn_bf   = (u16*)carve((size_t)Mrows * Dc * 2);
    u16* qkv_bf  = (u16*)carve((size_t)3 * Mrows * Dc * 2);     // [3][B][H][S][HD]
    u16* attn_bf = (u16*)carve((size_t)Mrows * Dc * 2);
    float* x1    = (float*)carve((size_t)Mrows * Dc * 4);
    u16* h_bf    = (u16*)carve((size_t)Mrows * Dc * 2);
    u16* act_bf  = (u16*)carve((size_t)Mrows * 4 * Dc * 2);

    auto tr = [&](const float* src, u16* dst, int Kdim, int Ndim) {
        transpose_bf16_kernel<<<dim3(Ndim / 32, Kdim / 32), 256, 0, stream>>>(src, dst, Kdim, Ndim);
    };
    tr(qkv_w,  qkvw_t,  Dc, 3 * Dc);
    tr(proj_w, projw_t, Dc, Dc);
    tr(mlp_w1, w1_t,    Dc, 4 * Dc);
    tr(mlp_w2, w2_t,    4 * Dc, Dc);

    const int mTiles = (Mrows + 127) / 128;   // 33

    // LN1 -> bf16
    ln_kernel<<<Mrows, 256, 0, stream>>>(x, ln1_w, ln1_b, xn_bf);

    // QKV GEMM + scatter (M=4100, N=2304, K=768)
    gemm_bf16_kernel<EP_QKV><<<dim3(3 * Dc / 128, mTiles), 256, 0, stream>>>(
        xn_bf, qkvw_t, qkv_b, nullptr, nullptr, qkv_bf, Mrows, 3 * Dc, Dc);

    // attention
    routed_attn_kernel<<<(Bc * Hc * Pc) / 8, 256, 0, stream>>>(qkv_bf, routes, attn_bf);
    cls_attn_kernel<<<Bc * Hc, 256, 0, stream>>>(qkv_bf, attn_bf);

    // proj GEMM + residual -> x1 (fp32)
    gemm_bf16_kernel<EP_RESID><<<dim3(Dc / 128, mTiles), 256, 0, stream>>>(
        attn_bf, projw_t, proj_b, x, x1, nullptr, Mrows, Dc, Dc);

    // LN2 -> bf16
    ln_kernel<<<Mrows, 256, 0, stream>>>(x1, ln2_w, ln2_b, h_bf);

    // MLP1 GEMM + GELU (M=4100, N=3072, K=768)
    gemm_bf16_kernel<EP_GELU><<<dim3(4 * Dc / 128, mTiles), 256, 0, stream>>>(
        h_bf, w1_t, mlp_b1, nullptr, nullptr, act_bf, Mrows, 4 * Dc, Dc);

    // MLP2 GEMM + residual -> out (M=4100, N=768, K=3072)
    gemm_bf16_kernel<EP_RESID><<<dim3(Dc / 128, mTiles), 256, 0, stream>>>(
        act_bf, w2_t, mlp_b2, x1, out, nullptr, Mrows, Dc, 4 * Dc);
}
